// ChebGCN_63282048139430
// MI455X (gfx1250) — compile-verified, matrix operand order checked by reading, not run on hardware
//
#include <hip/hip_runtime.h>
#include <hip/hip_bf16.h>

typedef __attribute__((ext_vector_type(16))) _Float16 v16h;
typedef __attribute__((ext_vector_type(8)))  float    v8f;

#define NNODES 100000
#define NEDGES 1600000
#define HFEAT  128

// ---------------------------------------------------------------------------
// degree: deg[row[e]] += w[e]
__global__ __launch_bounds__(256)
void deg_kernel(const int* __restrict__ row, const float* __restrict__ w,
                float* __restrict__ deg, int e) {
    int i = blockIdx.x * blockDim.x + threadIdx.x;
    if (i < e) atomicAdd(&deg[row[i]], w[i]);
}

// wn[e] = dinv[row]*w*dinv[col],  dinv = deg>0 ? rsqrt(deg) : 0
__global__ __launch_bounds__(256)
void wn_kernel(const int* __restrict__ row, const int* __restrict__ col,
               const float* __restrict__ w, const float* __restrict__ deg,
               float* __restrict__ wn, int e) {
    int i = blockIdx.x * blockDim.x + threadIdx.x;
    if (i >= e) return;
    float dr = deg[row[i]], dc = deg[col[i]];
    float ir = dr > 0.f ? rsqrtf(dr) : 0.f;
    float ic = dc > 0.f ? rsqrtf(dc) : 0.f;
    wn[i] = ir * w[i] * ic;
}

// ---------------------------------------------------------------------------
// out[row[e], :] += scale * wn[e] * v[col[e], :]   (out pre-zeroed)
// 32 lanes per edge, 4 features (float4) per lane. HBM-bound phase.
__global__ __launch_bounds__(256)
void scatter_lx(const float* __restrict__ v, const int* __restrict__ row,
                const int* __restrict__ col, const float* __restrict__ wn,
                float scale, float* __restrict__ out, int e) {
    int t  = blockIdx.x * blockDim.x + threadIdx.x;
    int ed = t >> 5;
    if (ed >= e) return;
    int f = (t & 31) * 4;
    float wv = wn[ed] * scale;
    const float4 x4 = *(const float4*)&v[(size_t)col[ed] * HFEAT + f];
    float* dst = &out[(size_t)row[ed] * HFEAT + f];
    atomicAdd(dst + 0, wv * x4.x);
    atomicAdd(dst + 1, wv * x4.y);
    atomicAdd(dst + 2, wv * x4.z);
    atomicAdd(dst + 3, wv * x4.w);
}

// t2 -= tx0 (elementwise, float4)
__global__ __launch_bounds__(256)
void axpy_sub(float* __restrict__ t2, const float* __restrict__ tx0, int n4) {
    int i = blockIdx.x * blockDim.x + threadIdx.x;
    if (i >= n4) return;
    float4 a = ((const float4*)t2)[i];
    float4 b = ((const float4*)tx0)[i];
    a.x -= b.x; a.y -= b.y; a.z -= b.z; a.w -= b.w;
    ((float4*)t2)[i] = a;
}

// ---------------------------------------------------------------------------
// D[M,128] = f( A[M,KD] @ W[KD,128] + Cin + bias )  via v_wmma_f32_16x16x32_f16.
// One wave32 owns TWO 16x16 output tiles stacked in M (32 rows): the strided B
// (weight) fragment is loaded once per K-step and reused by both WMMAs.
// Compile-time flags => straight-line epilogue, fully unrolled WMMA chain.
// Fragment layouts per CDNA5 ISA 7.12.2 (A 16x32 f16, B 32x16 f16, C/D f32).
template<int KD, bool HAS_C, bool HAS_BIAS, bool RELU>
__global__ __launch_bounds__(128)
void gemm_wmma(const float* __restrict__ A,
               const float* __restrict__ W,     // [KD,128] row-major
               const float* __restrict__ Cin,
               const float* __restrict__ bias,
               float* __restrict__ D,
               int mblocks) {
    const int lane = threadIdx.x;                 // 0..31 (wave32)
    const int wid  = blockIdx.x * 4 + threadIdx.y;
    if (wid >= mblocks * 8) return;               // uniform per wave
    const int mb = wid >> 3, nt = wid & 7;
    const int m0 = mb * 32, n0 = nt * 16;
    const int half = lane >> 4;
    const int l = lane & 15;
    const int NW = HFEAT;

    v8f c0 = {}, c1 = {};
    if constexpr (HAS_C) {
#pragma unroll
        for (int r = 0; r < 8; ++r) {
            c0[r] = Cin[(size_t)(m0 + r + half * 8) * NW + n0 + l];
            c1[r] = Cin[(size_t)(m0 + 16 + r + half * 8) * NW + n0 + l];
        }
    }

    const float* arow0 = A + (size_t)(m0 + l) * KD;
    const float* arow1 = A + (size_t)(m0 + 16 + l) * KD;
    const float* wcol  = W + n0 + l;

#pragma unroll
    for (int kk = 0; kk < KD; kk += 32) {
        const int ka = kk + half * 8;        // f16 slots 0..7
        const int kb = kk + 16 + half * 8;   // f16 slots 8..15

        v16h bv;
#pragma unroll
        for (int s = 0; s < 8; ++s) bv[s]     = (_Float16)wcol[(size_t)(ka + s) * NW];
#pragma unroll
        for (int s = 0; s < 8; ++s) bv[8 + s] = (_Float16)wcol[(size_t)(kb + s) * NW];

        float4 a0 = *(const float4*)(arow0 + ka);
        float4 a1 = *(const float4*)(arow0 + ka + 4);
        float4 a2 = *(const float4*)(arow0 + kb);
        float4 a3 = *(const float4*)(arow0 + kb + 4);
        v16h av0;
        av0[0]=(_Float16)a0.x;  av0[1]=(_Float16)a0.y;  av0[2]=(_Float16)a0.z;  av0[3]=(_Float16)a0.w;
        av0[4]=(_Float16)a1.x;  av0[5]=(_Float16)a1.y;  av0[6]=(_Float16)a1.z;  av0[7]=(_Float16)a1.w;
        av0[8]=(_Float16)a2.x;  av0[9]=(_Float16)a2.y;  av0[10]=(_Float16)a2.z; av0[11]=(_Float16)a2.w;
        av0[12]=(_Float16)a3.x; av0[13]=(_Float16)a3.y; av0[14]=(_Float16)a3.z; av0[15]=(_Float16)a3.w;

        float4 b0 = *(const float4*)(arow1 + ka);
        float4 b1 = *(const float4*)(arow1 + ka + 4);
        float4 b2 = *(const float4*)(arow1 + kb);
        float4 b3 = *(const float4*)(arow1 + kb + 4);
        v16h av1;
        av1[0]=(_Float16)b0.x;  av1[1]=(_Float16)b0.y;  av1[2]=(_Float16)b0.z;  av1[3]=(_Float16)b0.w;
        av1[4]=(_Float16)b1.x;  av1[5]=(_Float16)b1.y;  av1[6]=(_Float16)b1.z;  av1[7]=(_Float16)b1.w;
        av1[8]=(_Float16)b2.x;  av1[9]=(_Float16)b2.y;  av1[10]=(_Float16)b2.z; av1[11]=(_Float16)b2.w;
        av1[12]=(_Float16)b3.x; av1[13]=(_Float16)b3.y; av1[14]=(_Float16)b3.z; av1[15]=(_Float16)b3.w;

        c0 = __builtin_amdgcn_wmma_f32_16x16x32_f16(false, av0, false, bv,
                                                    (short)0, c0, false, false);
        c1 = __builtin_amdgcn_wmma_f32_16x16x32_f16(false, av1, false, bv,
                                                    (short)0, c1, false, false);
    }

    float bn = 0.f;
    if constexpr (HAS_BIAS) bn = bias[n0 + l];
#pragma unroll
    for (int r = 0; r < 8; ++r) {
        float v0 = c0[r], v1 = c1[r];
        if constexpr (HAS_BIAS) { v0 += bn; v1 += bn; }
        if constexpr (RELU) {
            v0 = v0 > 0.f ? v0 : 0.f;
            v1 = v1 > 0.f ? v1 : 0.f;
        }
        D[(size_t)(m0 + r + half * 8) * NW + n0 + l]      = v0;
        D[(size_t)(m0 + 16 + r + half * 8) * NW + n0 + l] = v1;
    }
}

// ---------------------------------------------------------------------------
// logits = h @ W2 + b2 (W2: [128,2]); 2-class softmax. One wave per node.
__global__ __launch_bounds__(256)
void head_softmax(const float* __restrict__ h, const float* __restrict__ W2,
                  const float* __restrict__ b2, float* __restrict__ out, int n) {
    int t = blockIdx.x * blockDim.x + threadIdx.x;
    int node = t >> 5;
    if (node >= n) return;
    int lane = t & 31;
    int f = lane * 4;
    float4 hv = *(const float4*)&h[(size_t)node * HFEAT + f];
    float p0 = hv.x * W2[(f+0)*2+0] + hv.y * W2[(f+1)*2+0] +
               hv.z * W2[(f+2)*2+0] + hv.w * W2[(f+3)*2+0];
    float p1 = hv.x * W2[(f+0)*2+1] + hv.y * W2[(f+1)*2+1] +
               hv.z * W2[(f+2)*2+1] + hv.w * W2[(f+3)*2+1];
#pragma unroll
    for (int off = 16; off >= 1; off >>= 1) {
        p0 += __shfl_xor(p0, off, 32);
        p1 += __shfl_xor(p1, off, 32);
    }
    if (lane == 0) {
        float l0 = p0 + b2[0], l1 = p1 + b2[1];
        float m = l0 > l1 ? l0 : l1;
        float e0 = __expf(l0 - m), e1 = __expf(l1 - m);
        float inv = 1.f / (e0 + e1);
        out[(size_t)node * 2 + 0] = e0 * inv;
        out[(size_t)node * 2 + 1] = e1 * inv;
    }
}

// ---------------------------------------------------------------------------
extern "C" void kernel_launch(void* const* d_in, const int* in_sizes, int n_in,
                              void* d_out, int out_size, void* d_ws, size_t ws_size,
                              hipStream_t stream) {
    const float* x   = (const float*)d_in[0];
    const int*   ei  = (const int*)  d_in[1];
    const float* ew  = (const float*)d_in[2];
    const float* W1  = (const float*)d_in[3];
    const float* b1  = (const float*)d_in[4];
    const float* c1W = (const float*)d_in[5];   // [3,128,128]
    const float* c1b = (const float*)d_in[6];
    const float* c2W = (const float*)d_in[7];   // [3,128,128]
    const float* c2b = (const float*)d_in[8];
    const float* W2  = (const float*)d_in[9];   // [128,2]
    const float* b2  = (const float*)d_in[10];
    float* out = (float*)d_out;

    const int Nn = NNODES, Ee = NEDGES;
    const int* row = ei;
    const int* col = ei + Ee;

    float* wsf = (float*)d_ws;
    float* wn  = wsf;                       // Ee
    float* deg = wn + Ee;                   // Nn
    float* B0  = deg + Nn;                  // Nn*128 each
    float* B1  = B0 + (size_t)Nn * HFEAT;
    float* B2  = B1 + (size_t)Nn * HFEAT;
    float* B3  = B2 + (size_t)Nn * HFEAT;
    const size_t nbBytes = (size_t)Nn * HFEAT * sizeof(float);

    // --- edge normalization ---
    hipMemsetAsync(deg, 0, Nn * sizeof(float), stream);
    deg_kernel<<<(Ee + 255) / 256, 256, 0, stream>>>(row, ew, deg, Ee);
    wn_kernel<<<(Ee + 255) / 256, 256, 0, stream>>>(row, col, ew, deg, wn, Ee);

    const int mblocks = Nn / 32;            // 3125, exact
    const dim3 gblk(32, 4);
    const int gemm_blocks = (mblocks * 8 + 3) / 4;
    const int scat_blocks = (Ee * 32) / 256;
    const int n4 = Nn * HFEAT / 4;

    // --- input projection: B0 = relu(x @ W1 + b1), KD = 256 ---
    gemm_wmma<256, false, true, true><<<gemm_blocks, gblk, 0, stream>>>(
        x, W1, nullptr, b1, B0, mblocks);

    // --- two ChebConv(K=3) layers; in = B0, dst = B0 ---
    const float* Wk[2] = { c1W, c2W };
    const float* bk[2] = { c1b, c2b };
    for (int layer = 0; layer < 2; ++layer) {
        const float* Wc = Wk[layer];
        // acc = Tx0 @ W[0]
        gemm_wmma<128, false, false, false><<<gemm_blocks, gblk, 0, stream>>>(
            B0, Wc, nullptr, nullptr, B3, mblocks);
        // Tx1 = L_hat @ Tx0
        hipMemsetAsync(B1, 0, nbBytes, stream);
        scatter_lx<<<scat_blocks, 256, 0, stream>>>(B0, row, col, wn, -1.f, B1, Ee);
        // acc += Tx1 @ W[1]
        gemm_wmma<128, true, false, false><<<gemm_blocks, gblk, 0, stream>>>(
            B1, Wc + 128 * 128, B3, nullptr, B3, mblocks);
        // Tx2 = 2 * L_hat @ Tx1 - Tx0
        hipMemsetAsync(B2, 0, nbBytes, stream);
        scatter_lx<<<scat_blocks, 256, 0, stream>>>(B1, row, col, wn, -2.f, B2, Ee);
        axpy_sub<<<(n4 + 255) / 256, 256, 0, stream>>>(B2, B0, n4);
        // dst = relu(acc + Tx2 @ W[2] + b)   (overwrites B0, Tx0 dead)
        gemm_wmma<128, true, true, true><<<gemm_blocks, gblk, 0, stream>>>(
            B2, Wc + 2 * 128 * 128, B3, bk[layer], B0, mblocks);
    }

    // --- head: softmax(h @ W2 + b2) ---
    head_softmax<<<(Nn * 32) / 256, 256, 0, stream>>>(B0, W2, b2, out, Nn);
}
